// BranchingQNetwork_67765993997197
// MI455X (gfx1250) — matrix operand order repaired and checked by reading, not compiled, not run on hardware
//
#include <hip/hip_runtime.h>
#include <stdint.h>

// ---------------- problem constants ----------------
#define B_    32
#define NS    2048          // sequence length N
#define OBS_  256
#define HID_  128
#define NEGC  9.0e15f

typedef __attribute__((ext_vector_type(16))) _Float16 v16h;
typedef __attribute__((ext_vector_type(8)))  _Float16 v8h;
typedef __attribute__((ext_vector_type(8)))  float    v8f;
typedef __attribute__((ext_vector_type(4)))  int      v4i;

// ---- CDNA5 async global->LDS path (probe via __has_builtin; sync fallback) ----
#if defined(__HIP_DEVICE_COMPILE__) && __has_builtin(__builtin_amdgcn_global_load_async_to_lds_b128)
#define HAVE_ASYNC_LDS 1
#endif

#if HAVE_ASYNC_LDS
__device__ __forceinline__ void async_cp16(const void* g, void* l) {
  typedef __attribute__((address_space(1))) v4i* gptr_t;   // global int4*
  typedef __attribute__((address_space(3))) v4i* lptr_t;   // LDS int4*
  __builtin_amdgcn_global_load_async_to_lds_b128(
      (gptr_t)(unsigned long long)(uintptr_t)g,
      (lptr_t)(unsigned int)(uintptr_t)l, 0, 0);
}
#if __has_builtin(__builtin_amdgcn_s_wait_asynccnt)
#define WAIT_ASYNC() __builtin_amdgcn_s_wait_asynccnt(0)
#else
#define WAIT_ASYNC() asm volatile("s_wait_asynccnt 0" ::: "memory")
#endif
#else
#define WAIT_ASYNC() ((void)0)
#endif

// D(f32) = A(16x32 f16) * B(32x16 f16) + C
__device__ __forceinline__ v8f wmma16(v16h a, v16h b, v8f c) {
  return __builtin_amdgcn_wmma_f32_16x16x32_f16(false, a, false, b, (short)0, c,
                                                false, false);
}

// A-fragment from its two contiguous 8-half groups
// (elem j: K = (lane>>4)*16 + j)
__device__ __forceinline__ v16h pack16(v8h lo, v8h hi) {
  v16h r;
#pragma unroll
  for (int i = 0; i < 8; ++i) { r[i] = lo[i]; r[i + 8] = hi[i]; }
  return r;
}

// ---------------- K0: transpose + convert weights to f16 ----------------
__global__ void convert_weights(const float* __restrict__ Wq, const float* __restrict__ Wk,
                                const float* __restrict__ Wv, const float* __restrict__ Wout,
                                _Float16* __restrict__ WqT, _Float16* __restrict__ WkT,
                                _Float16* __restrict__ WvT, _Float16* __restrict__ WoutT) {
  int tid = blockIdx.x * blockDim.x + threadIdx.x;
  const int PER = HID_ * OBS_;          // 32768
  if (tid < 3 * PER) {
    int m = tid / PER, r = tid % PER;
    int h = r / OBS_, k = r % OBS_;
    const float* src = (m == 0) ? Wq : (m == 1) ? Wk : Wv;
    _Float16*    dst = (m == 0) ? WqT : (m == 1) ? WkT : WvT;
    dst[h * OBS_ + k] = (_Float16)src[k * HID_ + h];
  } else {
    int r = tid - 3 * PER;
    if (r < HID_ * HID_) {
      int h = r / HID_, k = r % HID_;
      WoutT[h * HID_ + k] = (_Float16)Wout[k * HID_ + h];
    }
  }
}

// ---------------- K1: fused QKV projection (x read ONCE) ----------------------
__device__ __forceinline__ void proj_rowmajor(const v16h* A, const _Float16* __restrict__ WT,
                                              const float* __restrict__ bias,
                                              _Float16* __restrict__ dst,
                                              int rt, int half_id, int lm) {
#pragma unroll
  for (int ct = 0; ct < 8; ++ct) {
    v8f acc = {};
#pragma unroll
    for (int kc = 0; kc < 8; ++kc)
      acc = wmma16(A[kc],
                   *(const v16h*)(WT + (size_t)(ct * 16 + lm) * OBS_ + kc * 32 + half_id * 16),
                   acc);
    const float bn = bias[ct * 16 + lm];
#pragma unroll
    for (int r = 0; r < 8; ++r) {
      float v = acc[r] + bn; v = v > 0.f ? v : 0.f;
      dst[((size_t)rt * 16 + half_id * 8 + r) * HID_ + ct * 16 + lm] = (_Float16)v;
    }
  }
}

__device__ __forceinline__ void proj_vT(const v16h* A, const _Float16* __restrict__ WT,
                                        const float* __restrict__ bias,
                                        _Float16* __restrict__ vT,
                                        int rt, int half_id, int lm) {
#pragma unroll
  for (int ct = 0; ct < 8; ++ct) {
    v8f acc = {};
#pragma unroll
    for (int kc = 0; kc < 8; ++kc)
      acc = wmma16(A[kc],
                   *(const v16h*)(WT + (size_t)(ct * 16 + lm) * OBS_ + kc * 32 + half_id * 16),
                   acc);
    const float bn = bias[ct * 16 + lm];
    v8h pk;
#pragma unroll
    for (int r = 0; r < 8; ++r) {
      float v = acc[r] + bn; v = v > 0.f ? v : 0.f;
      pk[r] = (_Float16)v;
    }
    const int bId = (rt * 16) >> 11;            // NS == 2048
    const int nloc0 = (rt * 16) & (NS - 1);
    _Float16* dst = vT + ((size_t)bId * HID_ + ct * 16 + lm) * NS + nloc0 + half_id * 8;
    *(v8h*)dst = pk;                            // rows r consecutive -> 1x16B store
  }
}

__global__ __launch_bounds__(32) void qkv_kernel(
    const float* __restrict__ x,
    const _Float16* __restrict__ WqT, const _Float16* __restrict__ WkT,
    const _Float16* __restrict__ WvT,
    const float* __restrict__ bq, const float* __restrict__ bk, const float* __restrict__ bv,
    _Float16* __restrict__ q_h, _Float16* __restrict__ k_h, _Float16* __restrict__ vT) {
  const int rt = blockIdx.x;                    // 16-row tile over B*NS
  const int lane = threadIdx.x;
  const int half_id = lane >> 4, lm = lane & 15;

  const float* xrow = x + ((size_t)rt * 16 + lm) * OBS_;
  v16h A[8];
#pragma unroll
  for (int kc = 0; kc < 8; ++kc) {
    const float* p0 = xrow + kc * 32 + half_id * 16;
    float4 f0 = *(const float4*)(p0);
    float4 f1 = *(const float4*)(p0 + 4);
    float4 f2 = *(const float4*)(p0 + 8);
    float4 f3 = *(const float4*)(p0 + 12);
    float lo[8] = {f0.x, f0.y, f0.z, f0.w, f1.x, f1.y, f1.z, f1.w};
    float hi[8] = {f2.x, f2.y, f2.z, f2.w, f3.x, f3.y, f3.z, f3.w};
    v16h a;
#pragma unroll
    for (int i = 0; i < 8; ++i) { a[i] = (_Float16)lo[i]; a[i + 8] = (_Float16)hi[i]; }
    A[kc] = a;
  }
  proj_rowmajor(A, WqT, bq, q_h, rt, half_id, lm);
  proj_rowmajor(A, WkT, bk, k_h, rt, half_id, lm);
  proj_vT(A, WvT, bv, vT, rt, half_id, lm);
}

// ---------------- K2: flash attention --------------------------------------
// 4 waves / workgroup, each owns a 16-row q tile of the SAME batch; the
// workgroup shares double-buffered K (32x128) and V^T (128x32) f16 tiles in
// LDS, prefetched with async global->LDS (ASYNCcnt) when available.
__device__ __forceinline__ void stage_tiles(int nxt, int jt, int b, int tid,
                                            const _Float16* __restrict__ k_h,
                                            const _Float16* __restrict__ vT,
                                            _Float16 (*Ks)[32][HID_],
                                            _Float16 (*Vs)[HID_][32]) {
#pragma unroll
  for (int t = 0; t < 4; ++t) {
    int c = tid + t * 128;                       // 512 x 16B chunks per tile
    int krow = c >> 4, koff = (c & 15) * 8;
    const _Float16* gK = k_h + ((size_t)b * NS + jt * 32 + krow) * HID_ + koff;
    _Float16* lK = &Ks[nxt][krow][koff];
    int vh = c >> 2, voff = (c & 3) * 8;
    const _Float16* gV = vT + ((size_t)b * HID_ + vh) * NS + jt * 32 + voff;
    _Float16* lV = &Vs[nxt][vh][voff];
#if HAVE_ASYNC_LDS
    async_cp16(gK, lK);
    async_cp16(gV, lV);
#else
    *(v8h*)lK = *(const v8h*)gK;
    *(v8h*)lV = *(const v8h*)gV;
#endif
  }
}

__global__ __launch_bounds__(128) void attn_kernel(
    const _Float16* __restrict__ q_h, const _Float16* __restrict__ k_h,
    const _Float16* __restrict__ vT, const float* __restrict__ mask,
    _Float16* __restrict__ o_h) {
  __shared__ _Float16 Ks[2][32][HID_];           // 16 KB
  __shared__ _Float16 Vs[2][HID_][32];           // 16 KB
  __shared__ _Float16 Pst[4][16][32];            // 4 KB (per-wave P staging)

  const int tid = threadIdx.x;
  const int wid = tid >> 5, lane = tid & 31;
  const int half_id = lane >> 4, lm = lane & 15;
  const int b = blockIdx.x >> 5;                 // NS/64 == 32 groups / batch
  const int rt = (blockIdx.x & 31) * 4 + wid;    // this wave's 16-row q tile

  const _Float16* qrow = q_h + ((size_t)b * NS + rt * 16 + lm) * HID_;
  v16h aq[4];
#pragma unroll
  for (int kc = 0; kc < 4; ++kc)
    aq[kc] = pack16(*(const v8h*)(qrow + kc * 32 + half_id * 16),
                    *(const v8h*)(qrow + kc * 32 + 8 + half_id * 16));

  float mrow[8], lrow[8];
  v8f acc[8];
  const v8f vzero = {};
#pragma unroll
  for (int r = 0; r < 8; ++r) { mrow[r] = -3.0e38f; lrow[r] = 0.f; }
#pragma unroll
  for (int hc = 0; hc < 8; ++hc) acc[hc] = vzero;

  const float* maskBase = mask + ((size_t)b * NS + rt * 16) * NS;

  stage_tiles(0, 0, b, tid, k_h, vT, Ks, Vs);
  WAIT_ASYNC();
  __syncthreads();

  for (int jt = 0; jt < NS / 32; ++jt) {
    const int cur = jt & 1, nxt = cur ^ 1;
    if (jt + 1 < NS / 32)                        // prefetch next K/V tile
      stage_tiles(nxt, jt + 1, b, tid, k_h, vT, Ks, Vs);

    // ---- S = q . k^T for a 16x32 block (K=128 via 4 WMMAs per C-tile) ----
    v8f s[2];
#pragma unroll
    for (int nh = 0; nh < 2; ++nh) {
      v8f c = vzero;
#pragma unroll
      for (int kc = 0; kc < 4; ++kc)
        c = wmma16(aq[kc], *(const v16h*)(&Ks[cur][nh * 16 + lm][kc * 32 + half_id * 16]), c);
      s[nh] = c;
    }
    // ---- additive mask (streamed once -> non-temporal) ----
#pragma unroll
    for (int nh = 0; nh < 2; ++nh)
#pragma unroll
      for (int r = 0; r < 8; ++r) {
        float mv = __builtin_nontemporal_load(
            maskBase + (size_t)(half_id * 8 + r) * NS + jt * 32 + nh * 16 + lm);
        s[nh][r] = s[nh][r] * mv - NEGC * (1.0f - mv);
      }
    // ---- online softmax (row stats reduced across the 16-lane half) ----
    float mnew[8], scale[8];
#pragma unroll
    for (int r = 0; r < 8; ++r) {
      float t = fmaxf(s[0][r], s[1][r]);
#pragma unroll
      for (int off = 8; off; off >>= 1) t = fmaxf(t, __shfl_xor(t, off, 32));
      mnew[r] = fmaxf(mrow[r], t);
      scale[r] = __expf(mrow[r] - mnew[r]);
      mrow[r] = mnew[r];
    }
    float rs[8];
#pragma unroll
    for (int r = 0; r < 8; ++r) rs[r] = 0.f;
#pragma unroll
    for (int nh = 0; nh < 2; ++nh)
#pragma unroll
      for (int r = 0; r < 8; ++r) {
        float p = __expf(s[nh][r] - mnew[r]);
        rs[r] += p;
        Pst[wid][half_id * 8 + r][nh * 16 + lm] = (_Float16)p;
      }
#pragma unroll
    for (int r = 0; r < 8; ++r) {
      float t = rs[r];
#pragma unroll
      for (int off = 8; off; off >>= 1) t += __shfl_xor(t, off, 32);
      lrow[r] = lrow[r] * scale[r] + t;
#pragma unroll
      for (int hc = 0; hc < 8; ++hc) acc[hc][r] *= scale[r];
    }
    // D-layout -> A-layout relayout through per-wave LDS (in-wave DS order)
    __builtin_amdgcn_wave_barrier();
    v16h pa = pack16(*(const v8h*)(&Pst[wid][lm][half_id * 16]),
                     *(const v8h*)(&Pst[wid][lm][8 + half_id * 16]));
    __builtin_amdgcn_wave_barrier();
    // ---- acc += P(16x32) . V(32x128) : 8 WMMAs against LDS V tile ----
#pragma unroll
    for (int hc = 0; hc < 8; ++hc)
      acc[hc] = wmma16(pa, *(const v16h*)(&Vs[cur][hc * 16 + lm][half_id * 16]), acc[hc]);

    WAIT_ASYNC();                                // next tile landed (this wave)
    __syncthreads();                             // all waves: landed + done reading
  }
  // ---- normalize and emit f16 attention output ----
#pragma unroll
  for (int hc = 0; hc < 8; ++hc)
#pragma unroll
    for (int r = 0; r < 8; ++r) {
      size_t grow = (size_t)b * NS + rt * 16 + half_id * 8 + r;
      o_h[grow * HID_ + hc * 16 + lm] = (_Float16)(acc[hc][r] / lrow[r]);
    }
}

// ---------------- K3: out = relu(attn_out @ Wout + bout) ----------------------
__global__ __launch_bounds__(32) void outproj_kernel(
    const _Float16* __restrict__ o_h, const _Float16* __restrict__ WoutT,
    const float* __restrict__ bout, float* __restrict__ out2) {
  const int rt = blockIdx.x;
  const int lane = threadIdx.x;
  const int half_id = lane >> 4, lm = lane & 15;
  const _Float16* orow = o_h + ((size_t)rt * 16 + lm) * HID_;
  v16h A[4];
#pragma unroll
  for (int kc = 0; kc < 4; ++kc)
    A[kc] = pack16(*(const v8h*)(orow + kc * 32 + half_id * 16),
                   *(const v8h*)(orow + kc * 32 + 8 + half_id * 16));
#pragma unroll
  for (int ct = 0; ct < 8; ++ct) {
    v8f acc = {};
#pragma unroll
    for (int kc = 0; kc < 4; ++kc)
      acc = wmma16(A[kc],
                   *(const v16h*)(WoutT + (size_t)(ct * 16 + lm) * HID_ + kc * 32 + half_id * 16),
                   acc);
    const float bn = bout[ct * 16 + lm];
#pragma unroll
    for (int r = 0; r < 8; ++r) {
      float v = acc[r] + bn; v = v > 0.f ? v : 0.f;
      out2[((size_t)rt * 16 + half_id * 8 + r) * HID_ + ct * 16 + lm] = v;
    }
  }
}

// ---------------- K4: dueling heads (value / adv1 / adv2 per row) -------------
__global__ __launch_bounds__(32) void heads_kernel(
    const float* __restrict__ out2,
    const float* __restrict__ Wval, const float* __restrict__ bval,
    const float* __restrict__ Wadv1, const float* __restrict__ badv1,
    const float* __restrict__ Wadv2, const float* __restrict__ badv2,
    float* __restrict__ val, float* __restrict__ a1, float* __restrict__ a2) {
  const int lane = threadIdx.x;
  const size_t row0 = (size_t)blockIdx.x * 32;
  float wv[4], w1[4][8], w2[4][8];
#pragma unroll
  for (int i = 0; i < 4; ++i) {
    wv[i] = Wval[lane * 4 + i];
#pragma unroll
    for (int a = 0; a < 8; ++a) {
      w1[i][a] = Wadv1[(lane * 4 + i) * 8 + a];
      w2[i][a] = Wadv2[(lane * 4 + i) * 8 + a];
    }
  }
  const float bv0 = bval[0];
  for (int rr = 0; rr < 32; ++rr) {
    size_t row = row0 + rr;
    float4 o = *(const float4*)(out2 + row * HID_ + lane * 4);
    float o4[4] = {o.x, o.y, o.z, o.w};
    float vp = 0.f, p1[8], p2[8];
#pragma unroll
    for (int a = 0; a < 8; ++a) { p1[a] = 0.f; p2[a] = 0.f; }
#pragma unroll
    for (int i = 0; i < 4; ++i) {
      vp += o4[i] * wv[i];
#pragma unroll
      for (int a = 0; a < 8; ++a) { p1[a] += o4[i] * w1[i][a]; p2[a] += o4[i] * w2[i][a]; }
    }
#pragma unroll
    for (int off = 16; off; off >>= 1) {
      vp += __shfl_xor(vp, off, 32);
#pragma unroll
      for (int a = 0; a < 8; ++a) {
        p1[a] += __shfl_xor(p1[a], off, 32);
        p2[a] += __shfl_xor(p2[a], off, 32);
      }
    }
    if (lane == 0) {
      val[row] = vp + bv0;
#pragma unroll
      for (int a = 0; a < 8; ++a) {
        a1[row * 8 + a] = p1[a] + badv1[a];
        a2[row * 8 + a] = p2[a] + badv2[a];
      }
    }
  }
}

// ---------------- K5: deterministic per-batch head sums (for mean) ------------
__global__ void sums_kernel(const float* __restrict__ a1, const float* __restrict__ a2,
                            float* __restrict__ sum1, float* __restrict__ sum2) {
  __shared__ float red[256];
  const int b = blockIdx.x;
  const int ai = blockIdx.y;                    // 0..15
  const int a = ai & 7;
  const float* src = (ai < 8) ? a1 : a2;
  float*       dst = (ai < 8) ? sum1 : sum2;
  float p = 0.f;
  for (int n = threadIdx.x; n < NS; n += 256)
    p += src[((size_t)b * NS + n) * 8 + a];
  red[threadIdx.x] = p;
  __syncthreads();
  for (int s = 128; s; s >>= 1) {
    if ((int)threadIdx.x < s) red[threadIdx.x] += red[threadIdx.x + s];
    __syncthreads();
  }
  if (threadIdx.x == 0) dst[b * 8 + a] = red[0];
}

// ---------------- K6: q1/q2 = value + adv - mean(adv) -------------------------
__global__ void finalize_kernel(const float* __restrict__ val, const float* __restrict__ a1,
                                const float* __restrict__ a2, const float* __restrict__ sum1,
                                const float* __restrict__ sum2, float* __restrict__ out) {
  const int TOT = B_ * NS * 8;
  int idx = blockIdx.x * blockDim.x + threadIdx.x;
  if (idx >= TOT) return;
  int a = idx & 7;
  int row = idx >> 3;
  int b = row / NS;
  float v = val[row];
  const float inv = 1.0f / (float)NS;
  out[idx]       = v + a1[idx] - sum1[b * 8 + a] * inv;
  out[TOT + idx] = v + a2[idx] - sum2[b * 8 + a] * inv;
}

// ---------------- host side ---------------------------------------------------
extern "C" void kernel_launch(void* const* d_in, const int* in_sizes, int n_in,
                              void* d_out, int out_size, void* d_ws, size_t ws_size,
                              hipStream_t stream) {
  (void)in_sizes; (void)n_in; (void)out_size; (void)ws_size;
  const float* x     = (const float*)d_in[0];
  const float* mask  = (const float*)d_in[1];
  const float* Wv    = (const float*)d_in[2];
  const float* bv    = (const float*)d_in[3];
  const float* Wk    = (const float*)d_in[4];
  const float* bk    = (const float*)d_in[5];
  const float* Wq    = (const float*)d_in[6];
  const float* bq    = (const float*)d_in[7];
  const float* Wout  = (const float*)d_in[8];
  const float* bout  = (const float*)d_in[9];
  const float* Wval  = (const float*)d_in[10];
  const float* bval  = (const float*)d_in[11];
  const float* Wadv1 = (const float*)d_in[12];
  const float* badv1 = (const float*)d_in[13];
  const float* Wadv2 = (const float*)d_in[14];
  const float* badv2 = (const float*)d_in[15];

  char* ws = (char*)d_ws;
  constexpr size_t OFF_WQT  = 0;                    // 64 KB each
  constexpr size_t OFF_WKT  = 65536;
  constexpr size_t OFF_WVT  = 131072;
  constexpr size_t OFF_WOT  = 196608;               // 32 KB
  constexpr size_t OFF_QH   = 262144;               // 16 MB f16 [B*NS][HID]
  constexpr size_t OFF_KH   = OFF_QH  + 16777216;
  constexpr size_t OFF_VT   = OFF_KH  + 16777216;   // f16 [B][HID][NS]
  constexpr size_t OFF_OH   = OFF_VT  + 16777216;   // f16 attn out
  constexpr size_t OFF_OUT2 = OFF_OH  + 16777216;   // f32 [B*NS][HID] (32 MB)
  constexpr size_t OFF_VAL  = OFF_OUT2 + 33554432;
  constexpr size_t OFF_A1   = OFF_VAL + 262144;
  constexpr size_t OFF_A2   = OFF_A1  + 2097152;
  constexpr size_t OFF_S1   = OFF_A2  + 2097152;
  constexpr size_t OFF_S2   = OFF_S1  + 1024;       // total ~105.4 MB

  _Float16* WqT   = (_Float16*)(ws + OFF_WQT);
  _Float16* WkT   = (_Float16*)(ws + OFF_WKT);
  _Float16* WvT   = (_Float16*)(ws + OFF_WVT);
  _Float16* WoutT = (_Float16*)(ws + OFF_WOT);
  _Float16* q_h   = (_Float16*)(ws + OFF_QH);
  _Float16* k_h   = (_Float16*)(ws + OFF_KH);
  _Float16* vT    = (_Float16*)(ws + OFF_VT);
  _Float16* o_h   = (_Float16*)(ws + OFF_OH);
  float*    out2  = (float*)(ws + OFF_OUT2);
  float*    val   = (float*)(ws + OFF_VAL);
  float*    a1    = (float*)(ws + OFF_A1);
  float*    a2    = (float*)(ws + OFF_A2);
  float*    sum1  = (float*)(ws + OFF_S1);
  float*    sum2  = (float*)(ws + OFF_S2);

  convert_weights<<<dim3(448), 256, 0, stream>>>(Wq, Wk, Wv, Wout, WqT, WkT, WvT, WoutT);
  qkv_kernel<<<dim3((B_ * NS) / 16), 32, 0, stream>>>(x, WqT, WkT, WvT, bq, bk, bv,
                                                      q_h, k_h, vT);
  attn_kernel<<<dim3(B_ * (NS / 64)), 128, 0, stream>>>(q_h, k_h, vT, mask, o_h);
  outproj_kernel<<<dim3((B_ * NS) / 16), 32, 0, stream>>>(o_h, WoutT, bout, out2);
  heads_kernel<<<dim3((B_ * NS) / 32), 32, 0, stream>>>(out2, Wval, bval, Wadv1, badv1,
                                                        Wadv2, badv2, val, a1, a2);
  sums_kernel<<<dim3(B_, 16), 256, 0, stream>>>(a1, a2, sum1, sum2);
  finalize_kernel<<<dim3(2048), 256, 0, stream>>>(val, a1, a2, sum1, sum2, (float*)d_out);
}
